// MixtureOfDepthsRouter_50929722196043
// MI455X (gfx1250) — matrix operand order repaired and checked by reading, not compiled, not run on hardware
//
#include <hip/hip_runtime.h>
#include <hip/hip_bf16.h>

#define DIM   4096
#define HID   1024
#define BATCH 4
#define SEQ   4096
#define MROWS (BATCH * SEQ)   // 16384
#define KSEL  (SEQ / 2)       // 2048

#define BM 128
#define BN 128
#define BK 64
#define NKT (DIM / BK)        // 64 K-chunks
#define XF_PITCH 68           // f32 per xf row (64 + 4 pad; rows stay 16B-aligned)
#define WPITCH   40           // u32 per ws row (32 + 8 pad; rows stay 16B-aligned)

#if defined(__has_builtin)
# if __has_builtin(__builtin_amdgcn_global_load_async_to_lds_b128)
#  define ASYNC_LDS 1
# endif
#endif
#ifndef ASYNC_LDS
# define ASYNC_LDS 0
#endif

typedef __attribute__((ext_vector_type(16))) __bf16       v16bf;
typedef __attribute__((ext_vector_type(8)))  float        v8f;
typedef __attribute__((ext_vector_type(8)))  unsigned int v8u;
typedef __attribute__((ext_vector_type(4)))  int          v4i;

union Frag { v16bf bf; v8u u; };

__device__ __forceinline__ unsigned int f2bf(float f) {
    unsigned int u = __float_as_uint(f);
    u += 0x7fffu + ((u >> 16) & 1u);   // round-to-nearest-even
    return u >> 16;
}

__device__ __forceinline__ unsigned int pack_bf16x2(float2 v) {
    union { __hip_bfloat162 h; unsigned int u; } cv;
    cv.h = __float22bfloat162_rn(v);   // v_cvt_pk_bf16_f32
    return cv.u;
}

// 16B async DMA copy global -> LDS (ASYNCcnt); falls back to a plain copy.
__device__ __forceinline__ void async_copy_b128(const void* g, void* l) {
#if ASYNC_LDS
    typedef __attribute__((address_space(1))) v4i* gp_t;   // global int4*
    typedef __attribute__((address_space(3))) v4i* lp_t;   // LDS int4*
    __builtin_amdgcn_global_load_async_to_lds_b128((gp_t)(void*)g, (lp_t)l, 0, 0);
#else
    *(uint4*)l = *(const uint4*)g;
#endif
}

__device__ __forceinline__ void wait_async() {
#if ASYNC_LDS
# if __has_builtin(__builtin_amdgcn_s_wait_asynccnt)
    __builtin_amdgcn_s_wait_asynccnt(0);
# else
    asm volatile("s_wait_asynccnt 0x0" ::: "memory");
# endif
#endif
}

// ---------------------------------------------------------------------------
// Kernel 1: transpose+convert w1 [DIM][HID] f32  ->  w1p [HID][DIM] bf16
// ---------------------------------------------------------------------------
__global__ __launch_bounds__(256)
void pack_w1_kernel(const float* __restrict__ w1, unsigned short* __restrict__ w1p) {
    __shared__ float tile[32][33];
    const int k0 = blockIdx.y * 32;
    const int n0 = blockIdx.x * 32;
    const int tx = threadIdx.x;          // 0..31
    const int ty = threadIdx.y;          // 0..7
    #pragma unroll
    for (int i = ty; i < 32; i += 8)
        tile[i][tx] = w1[(size_t)(k0 + i) * HID + n0 + tx];
    __syncthreads();
    #pragma unroll
    for (int i = ty; i < 32; i += 8)
        w1p[(size_t)(n0 + i) * DIM + k0 + tx] = (unsigned short)f2bf(tile[tx][i]);
}

// ---------------------------------------------------------------------------
// Kernel 2: fused scorer  partial[nb][m] = sum_{n in nb} relu(x@w1 + b1) * w2
// bf16 WMMA, f32 accumulate.  Async-DMA double-buffered LDS tiles (no staging
// registers, DMA overlaps WMMA) + double-buffered B register fragments.
// kt loop unrolled x2 so the two LDS buffers get disjoint register contexts
// (removes WMMA->VALU WAR-hazard v_nops on the A-fragment registers).
// ---------------------------------------------------------------------------
__global__ __launch_bounds__(256)
void mlp_score_kernel(const float* __restrict__ x,
                      const unsigned short* __restrict__ w1p,
                      const float* __restrict__ b1,
                      const float* __restrict__ w2,
                      float* __restrict__ partial)   // [HID/BN][MROWS]
{
    __shared__ float        xf[2][BM * XF_PITCH];   // raw f32 x tile
    __shared__ unsigned int ws[2][BN * WPITCH];     // bf16 w tile

    const int tid  = threadIdx.x;
    const int wave = tid >> 5;
    const int lane = tid & 31;
    const int half = lane >> 4;
    const int ln   = lane & 15;

    const int mBase = blockIdx.y * BM;
    const int nBase = blockIdx.x * BN;

    v8f acc[8];
    #pragma unroll
    for (int nt = 0; nt < 8; ++nt)
        #pragma unroll
        for (int i = 0; i < 8; ++i) acc[nt][i] = 0.0f;

    auto issue_copy = [&](int buf, int kt) {
        const int kk = kt * BK;
        // x tile: BM x BK f32 = 32KB, 16B per lane per op, 8 ops/thread
        #pragma unroll
        for (int i = 0; i < 8; ++i) {
            int flat = tid + i * 256;          // 0..2047
            int row  = flat >> 4;              // 16 x 16B chunks per row
            int ch   = flat & 15;
            async_copy_b128(x + (size_t)(mBase + row) * DIM + kk + ch * 4,
                            &xf[buf][row * XF_PITCH + ch * 4]);
        }
        // w tile: BN x BK bf16 = 16KB, 4 ops/thread
        #pragma unroll
        for (int i = 0; i < 4; ++i) {
            int flat = tid + i * 256;          // 0..1023
            int row  = flat >> 3;              // 8 x 16B chunks per row
            int ch   = flat & 7;
            async_copy_b128(w1p + (size_t)(nBase + row) * DIM + kk + ch * 8,
                            &ws[buf][row * WPITCH + ch * 4]);
        }
    };

    auto compute_chunk = [&](int buf) {
        Frag A[2], Bf[2];
        const float2* arow = (const float2*)&xf[buf][(wave * 16 + ln) * XF_PITCH];
        #pragma unroll
        for (int v = 0; v < 8; ++v) {
            const int off = v + half * 8;  // float2 units
            A[0].u[v] = pack_bf16x2(arow[off]);
            A[1].u[v] = pack_bf16x2(arow[16 + off]);
        }
        {   // prime B pipeline: step 0
            const unsigned int* p = &ws[buf][ln * WPITCH + half * 8];
            #pragma unroll
            for (int v = 0; v < 8; ++v) Bf[0].u[v] = p[v];
        }
        #pragma unroll
        for (int s = 0; s < 16; ++s) {
            const int ks = s >> 3, nt = s & 7;
            if (s < 15) {                      // preload step s+1 into other buffer
                const int ns = s + 1, nks = ns >> 3, nnt = ns & 7;
                const unsigned int* p =
                    &ws[buf][(nnt * 16 + ln) * WPITCH + nks * 16 + half * 8];
                #pragma unroll
                for (int v = 0; v < 8; ++v) Bf[ns & 1].u[v] = p[v];
            }
            acc[nt] = __builtin_amdgcn_wmma_f32_16x16x32_bf16(
                false, A[ks].bf, false, Bf[s & 1].bf, (short)0, acc[nt], false, false);
        }
    };

    issue_copy(0, 0);
    wait_async();

    #pragma unroll 2
    for (int kt = 0; kt < NKT; ++kt) {
        const int buf = kt & 1;
        __syncthreads();                       // chunk kt ready; buf^1 reads done
        if (kt + 1 < NKT) issue_copy(buf ^ 1, kt + 1);   // DMA overlaps compute
        compute_chunk(buf);
        wait_async();                          // chunk kt+1 DMA done before barrier
    }

    // Epilogue: +b1, relu, dot with w2, reduce across the 16 lanes of each half
    float part[8];
    #pragma unroll
    for (int i = 0; i < 8; ++i) part[i] = 0.0f;
    #pragma unroll
    for (int nt = 0; nt < 8; ++nt) {
        const int n    = nBase + nt * 16 + ln;
        const float wv = w2[n];
        const float bv = b1[n];
        #pragma unroll
        for (int i = 0; i < 8; ++i) {
            float h = acc[nt][i] + bv;
            h = fmaxf(h, 0.0f);
            part[i] = fmaf(h, wv, part[i]);
        }
    }
    #pragma unroll
    for (int off = 1; off < 16; off <<= 1)
        #pragma unroll
        for (int i = 0; i < 8; ++i) part[i] += __shfl_xor(part[i], off, 32);

    if (ln == 0) {
        const int mrow = mBase + wave * 16 + half * 8;   // C layout: VGPR i -> M = i + 8*half
        #pragma unroll
        for (int i = 0; i < 8; ++i)
            partial[(size_t)blockIdx.x * MROWS + mrow + i] = part[i];
    }
}

// ---------------------------------------------------------------------------
// Kernel 3: per-batch-row top-k router (sort 4096, softmax, index-sort 2048)
// ---------------------------------------------------------------------------
__global__ __launch_bounds__(1024)
void router_kernel(const float* __restrict__ partial,  // [HID/BN][MROWS]
                   const float* __restrict__ b2,
                   float* __restrict__ mask_out,
                   float* __restrict__ rw_out)
{
    __shared__ float key[SEQ];
    __shared__ int   idx[SEQ];
    __shared__ int   cidx[KSEL];
    __shared__ float red[1024];

    const int b    = blockIdx.x;
    const int tid  = threadIdx.x;
    const float bias = b2[0];

    for (int s = tid; s < SEQ; s += 1024) {
        float v = 0.0f;
        #pragma unroll
        for (int p = 0; p < HID / BN; ++p)
            v += partial[(size_t)p * MROWS + b * SEQ + s];
        key[s] = v + bias;        // TEMPERATURE == 1
        idx[s] = s;
    }
    __syncthreads();

    // bitonic sort: descending score, ascending index on ties (matches top_k)
    for (int k = 2; k <= SEQ; k <<= 1) {
        for (int j = k >> 1; j > 0; j >>= 1) {
            for (int t = tid; t < SEQ; t += 1024) {
                int p = t ^ j;
                if (p > t) {
                    bool up = ((t & k) == 0);
                    float ka = key[t], kb = key[p];
                    int   ia = idx[t], ib = idx[p];
                    bool inOrder = (ka > kb) || (ka == kb && ia < ib);
                    if (up != inOrder) { key[t] = kb; key[p] = ka; idx[t] = ib; idx[p] = ia; }
                }
            }
            __syncthreads();
        }
    }

    // mask: idx[] is a permutation of [0,SEQ) -> one racefree pass
    for (int t = tid; t < SEQ; t += 1024)
        mask_out[b * SEQ + idx[t]] = (t < KSEL) ? 1.0f : 0.0f;

    // softmax over the top KSEL scores (descending => max = key[0])
    const float mx = key[0];
    float loc = 0.0f;
    for (int t = tid; t < KSEL; t += 1024) loc += __expf(key[t] - mx);
    red[tid] = loc;
    __syncthreads();
    for (int s = 512; s > 0; s >>= 1) {
        if (tid < s) red[tid] += red[tid + s];
        __syncthreads();
    }
    const float rs = 1.0f / red[0];

    // keep a copy of the top-k indices, zero the non-selected routing slots
    for (int t = tid; t < KSEL; t += 1024) cidx[t] = idx[t];
    for (int t = tid; t < SEQ; t += 1024)
        if (t >= KSEL) rw_out[b * SEQ + idx[t]] = 0.0f;
    for (int t = tid; t < KSEL; t += 1024)
        key[t] = __expf(key[t] - mx) * rs;    // soft[j] in top-k order
    __syncthreads();

    // sort the top-k indices ascending (values stay positional, per reference)
    for (int k = 2; k <= KSEL; k <<= 1) {
        for (int j = k >> 1; j > 0; j >>= 1) {
            for (int t = tid; t < KSEL; t += 1024) {
                int p = t ^ j;
                if (p > t) {
                    bool up = ((t & k) == 0);
                    int a = cidx[t], c = cidx[p];
                    bool inOrder = (a < c);
                    if (up != inOrder) { cidx[t] = c; cidx[p] = a; }
                }
            }
            __syncthreads();
        }
    }

    // routing_weights[sorted_idx[j]] = soft[j]
    for (int j = tid; j < KSEL; j += 1024)
        rw_out[b * SEQ + cidx[j]] = key[j];
}

// ---------------------------------------------------------------------------
extern "C" void kernel_launch(void* const* d_in, const int* in_sizes, int n_in,
                              void* d_out, int out_size, void* d_ws, size_t ws_size,
                              hipStream_t stream)
{
    const float* x  = (const float*)d_in[0];
    const float* w1 = (const float*)d_in[1];
    const float* b1 = (const float*)d_in[2];
    const float* w2 = (const float*)d_in[3];
    const float* b2 = (const float*)d_in[4];

    unsigned short* w1p   = (unsigned short*)d_ws;                         // 8 MB bf16
    float*          parts = (float*)((char*)d_ws + (size_t)HID * DIM * 2); // 8*16384 f32

    dim3 tb(32, 8);
    dim3 tg(HID / 32, DIM / 32);
    pack_w1_kernel<<<tg, tb, 0, stream>>>(w1, w1p);

    dim3 gg(HID / BN, MROWS / BM);   // (8, 128): nBlock fastest -> x tiles reused in L2
    mlp_score_kernel<<<gg, 256, 0, stream>>>(x, w1p, b1, w2, parts);

    float* mask_out = (float*)d_out;
    float* rw_out   = mask_out + (size_t)BATCH * SEQ;
    router_kernel<<<BATCH, 1024, 0, stream>>>(parts, b2, mask_out, rw_out);
}
